// ScaledDotProductAttention_51668456570989
// MI455X (gfx1250) — matrix operand order repaired and checked by reading, not compiled, not run on hardware
//
#include <hip/hip_runtime.h>
#include <math.h>

typedef __attribute__((ext_vector_type(16))) __bf16 v16bf;
typedef __attribute__((ext_vector_type(8)))  __bf16 v8bf;
typedef __attribute__((ext_vector_type(4)))  __bf16 v4bf;
typedef __attribute__((ext_vector_type(8)))  float  v8f;

namespace {
constexpr int B_ = 16, S_ = 2048, D_ = 128;
constexpr int QT = 16;            // query rows per wave
constexpr int KT = 64;            // key rows per tile iteration
constexpr int NWAVE = 8;          // waves per workgroup
constexpr int QB = QT * NWAVE;    // 128 query rows per workgroup

union V16 { v16bf v; v8bf h[2]; };
} // namespace

// xor-shuffle within each 16-lane half via ds_swizzle (group-of-32 mode:
// offset[15]=0, xor=offset[14:10], and_mask=0x1f). Masks 1/2/4/8 stay in-half.
#define SWZ_XOR_F32(x, m) \
  __int_as_float(__builtin_amdgcn_ds_swizzle(__float_as_int(x), (0x1f | ((m) << 10))))

__global__ __launch_bounds__(256)
void fattn_wmma_bf16(const float* __restrict__ Q, const float* __restrict__ K,
                     const float* __restrict__ V, float* __restrict__ Out)
{
  // double-buffered K/V tiles: one barrier per iteration, fill(next) overlaps
  // compute(cur) at block level (only the filling wave stalls on loadcnt)
  __shared__ __align__(16) __bf16 sK[2][KT * D_];     // K tile, row-major (64x128)
  __shared__ __align__(16) __bf16 sVT[2][D_ * KT];    // V tile, transposed (128x64)
  __shared__ __align__(16) __bf16 sP[NWAVE][QT * KT]; // per-wave P staging (16x64)

  const int tid  = threadIdx.x;
  const int wave = tid >> 5;
  const int lane = tid & 31;
  const int lm   = lane & 15;   // lane % 16
  const int lh   = lane >> 4;   // lane half (0/1)

  const int nQB = S_ / QB;
  const int b   = blockIdx.x / nQB;
  const int q0  = (blockIdx.x % nQB) * QB + wave * QT;

  const float* kbase = K + (size_t)b * S_ * D_;
  const float* vbase = V + (size_t)b * S_ * D_;

  // cooperative fp32->bf16 tile fill into buffer `buf` for keys [kbb, kbb+KT)
  auto fill = [&](int buf, int kbb) {
    // sK: 4 threads/row, 32 contiguous cols -> four ds_store_b128 per thread
    const int row = tid >> 2;
    const int c0  = (tid & 3) * 32;
    const float* ks = kbase + (size_t)(kbb + row) * D_ + c0;
    #pragma unroll
    for (int half = 0; half < 2; ++half) {
      V16 kk;
      #pragma unroll
      for (int i = 0; i < 16; i += 4) {
        float4 f = *(const float4*)(ks + 16 * half + i);
        kk.v[i + 0] = (__bf16)f.x;
        kk.v[i + 1] = (__bf16)f.y;
        kk.v[i + 2] = (__bf16)f.z;
        kk.v[i + 3] = (__bf16)f.w;
      }
      *(v8bf*)&sK[buf][row * D_ + c0 + 16 * half]     = kk.h[0];
      *(v8bf*)&sK[buf][row * D_ + c0 + 16 * half + 8] = kk.h[1];
    }
    // sVT: two 4x4 block transposes -> eight ds_store_b64 per thread
    const int kq = (tid >> 5) * 8;   // 0..56
    const int dq = (tid & 31) * 4;   // 0..124
    #pragma unroll
    for (int g = 0; g < 2; ++g) {
      const float* vs = vbase + (size_t)(kbb + kq + 4 * g) * D_ + dq;
      float4 r0 = *(const float4*)(vs);
      float4 r1 = *(const float4*)(vs + D_);
      float4 r2 = *(const float4*)(vs + 2 * D_);
      float4 r3 = *(const float4*)(vs + 3 * D_);
      v4bf c0v, c1v, c2v, c3v;
      c0v[0] = (__bf16)r0.x; c0v[1] = (__bf16)r1.x; c0v[2] = (__bf16)r2.x; c0v[3] = (__bf16)r3.x;
      c1v[0] = (__bf16)r0.y; c1v[1] = (__bf16)r1.y; c1v[2] = (__bf16)r2.y; c1v[3] = (__bf16)r3.y;
      c2v[0] = (__bf16)r0.z; c2v[1] = (__bf16)r1.z; c2v[2] = (__bf16)r2.z; c2v[3] = (__bf16)r3.z;
      c3v[0] = (__bf16)r0.w; c3v[1] = (__bf16)r1.w; c3v[2] = (__bf16)r2.w; c3v[3] = (__bf16)r3.w;
      *(v4bf*)&sVT[buf][(dq + 0) * KT + kq + 4 * g] = c0v;
      *(v4bf*)&sVT[buf][(dq + 1) * KT + kq + 4 * g] = c1v;
      *(v4bf*)&sVT[buf][(dq + 2) * KT + kq + 4 * g] = c2v;
      *(v4bf*)&sVT[buf][(dq + 3) * KT + kq + 4 * g] = c3v;
    }
  };

  // 1/sqrt(D) * log2(e): softmax in base-2 domain, scale folded into Q
  const float qscale = 0.08838834764831845f * 1.4426950408889634f;

  // ---- Q fragments in A-matrix layout (lane=row; K runs contiguous {16h..16h+15}) ----
  V16 aq[4];
  {
    const float* qp = Q + ((size_t)b * S_ + (q0 + lm)) * (size_t)D_;
    #pragma unroll
    for (int c = 0; c < 4; ++c) {
      #pragma unroll
      for (int run = 0; run < 2; ++run) {
        const int col = 32 * c + 16 * lh + 8 * run;
        float4 f0 = *(const float4*)(qp + col);
        float4 f1 = *(const float4*)(qp + col + 4);
        const int j = 8 * run;
        aq[c].v[j + 0] = (__bf16)(f0.x * qscale);
        aq[c].v[j + 1] = (__bf16)(f0.y * qscale);
        aq[c].v[j + 2] = (__bf16)(f0.z * qscale);
        aq[c].v[j + 3] = (__bf16)(f0.w * qscale);
        aq[c].v[j + 4] = (__bf16)(f1.x * qscale);
        aq[c].v[j + 5] = (__bf16)(f1.y * qscale);
        aq[c].v[j + 6] = (__bf16)(f1.z * qscale);
        aq[c].v[j + 7] = (__bf16)(f1.w * qscale);
      }
    }
  }

  // all-ones B fragment: row-sum of P via WMMA (sums replicated across lanes)
  V16 ones;
  #pragma unroll
  for (int j = 0; j < 16; ++j) ones.v[j] = (__bf16)1.0f;

  float m[8], l[8];
  v8f o[8];
  #pragma unroll
  for (int r = 0; r < 8; ++r) { m[r] = -INFINITY; l[r] = 0.f; }
  #pragma unroll
  for (int dt = 0; dt < 8; ++dt) o[dt] = {};

  fill(0, 0);   // prime first tile

  for (int kb = 0, it = 0; kb < S_; kb += KT, ++it) {
    const int cur = it & 1;
    __syncthreads();   // fill(cur) visible; previous compute on (cur^1) done
    if (kb + KT < S_) {
      if (kb + 2 * KT < S_) {  // prefetch the tile after next (global_prefetch_b8)
        __builtin_prefetch(kbase + (size_t)(kb + 2 * KT) * D_ + tid * 32, 0, 1);
        __builtin_prefetch(vbase + (size_t)(kb + 2 * KT) * D_ + tid * 32, 0, 1);
      }
      fill(cur ^ 1, kb + KT);
    }

    // ---- scores: S = Q' K^T, four 16x16 tiles (keys 16t+lm as columns) ----
    // B layout: lane = key column; per-lane contiguous 16 d-values @ 16*lh
    v8f s[4];
    #pragma unroll
    for (int t = 0; t < 4; ++t) {
      s[t] = {};
      V16 bk[4];
      #pragma unroll
      for (int c = 0; c < 4; ++c)
        bk[c].v = *(const v16bf*)&sK[cur][(16 * t + lm) * D_ + 32 * c + 16 * lh];
      #pragma unroll
      for (int c = 0; c < 4; ++c)
        s[t] = __builtin_amdgcn_wmma_f32_16x16x32_bf16(false, aq[c].v, false, bk[c].v,
                                                       (short)0, s[t], false, false);
    }

    // ---- online softmax: row max via ds_swizzle xor-reduce; write P to LDS ----
    float corr[8];
    #pragma unroll
    for (int r = 0; r < 8; ++r) {
      float bm = fmaxf(fmaxf(s[0][r], s[1][r]), fmaxf(s[2][r], s[3][r]));
      bm = fmaxf(bm, SWZ_XOR_F32(bm, 1));
      bm = fmaxf(bm, SWZ_XOR_F32(bm, 2));
      bm = fmaxf(bm, SWZ_XOR_F32(bm, 4));
      bm = fmaxf(bm, SWZ_XOR_F32(bm, 8));
      const float mn = fmaxf(m[r], bm);
      corr[r] = exp2f(m[r] - mn);
      m[r] = mn;
      const int row = r + 8 * lh;
      sP[wave][row * KT + lm]      = (__bf16)exp2f(s[0][r] - mn);
      sP[wave][row * KT + 16 + lm] = (__bf16)exp2f(s[1][r] - mn);
      sP[wave][row * KT + 32 + lm] = (__bf16)exp2f(s[2][r] - mn);
      sP[wave][row * KT + 48 + lm] = (__bf16)exp2f(s[3][r] - mn);
    }
    #pragma unroll
    for (int dt = 0; dt < 8; ++dt) {
      #pragma unroll
      for (int r = 0; r < 8; ++r) o[dt][r] *= corr[r];
    }

    // same-wave DS ops are in-order: reads below observe the stores above
    V16 ap0, ap1;   // P in A layout: keys 0..31 and 32..63
    ap0.h[0] = *(const v8bf*)&sP[wave][lm * KT + 16 * lh];
    ap0.h[1] = *(const v8bf*)&sP[wave][lm * KT + 8 + 16 * lh];
    ap1.h[0] = *(const v8bf*)&sP[wave][lm * KT + 32 + 16 * lh];
    ap1.h[1] = *(const v8bf*)&sP[wave][lm * KT + 40 + 16 * lh];

    // ---- row sums of P via WMMA with all-ones B; update l ----
    {
      v8f z = {};
      v8f rs = __builtin_amdgcn_wmma_f32_16x16x32_bf16(false, ap0.v, false, ones.v,
                                                       (short)0, z, false, false);
      rs = __builtin_amdgcn_wmma_f32_16x16x32_bf16(false, ap1.v, false, ones.v,
                                                   (short)0, rs, false, false);
      #pragma unroll
      for (int r = 0; r < 8; ++r) l[r] = l[r] * corr[r] + rs[r];
    }

    // ---- O += P V : B operand = V columns (lane = d column) ----
    #pragma unroll
    for (int dt = 0; dt < 8; ++dt) {
      V16 bv0, bv1;
      bv0.v = *(const v16bf*)&sVT[cur][(dt * 16 + lm) * KT + 16 * lh];
      bv1.v = *(const v16bf*)&sVT[cur][(dt * 16 + lm) * KT + 32 + 16 * lh];
      o[dt] = __builtin_amdgcn_wmma_f32_16x16x32_bf16(false, ap0.v, false, bv0.v,
                                                      (short)0, o[dt], false, false);
      o[dt] = __builtin_amdgcn_wmma_f32_16x16x32_bf16(false, ap1.v, false, bv1.v,
                                                      (short)0, o[dt], false, false);
    }
  }

  // ---- epilogue: normalize by l and store fp32 ----
  #pragma unroll
  for (int r = 0; r < 8; ++r) {
    const int row = q0 + r + 8 * lh;
    const float inv = 1.f / l[r];
    float* op = Out + ((size_t)b * S_ + row) * (size_t)D_;
    #pragma unroll
    for (int dt = 0; dt < 8; ++dt)
      op[dt * 16 + lm] = o[dt][r] * inv;
  }
}

extern "C" void kernel_launch(void* const* d_in, const int* in_sizes, int n_in,
                              void* d_out, int out_size, void* d_ws, size_t ws_size,
                              hipStream_t stream) {
  (void)in_sizes; (void)n_in; (void)out_size; (void)d_ws; (void)ws_size;
  const float* Q = (const float*)d_in[0];
  const float* K = (const float*)d_in[1];
  const float* V = (const float*)d_in[2];
  float* O = (float*)d_out;
  dim3 grid(B_ * (S_ / QB));   // 16 batches * 16 query-blocks = 256 workgroups
  fattn_wmma_bf16<<<grid, dim3(256), 0, stream>>>(Q, K, V, O);
}
